// RNNModel_5385888989382
// MI455X (gfx1250) — matrix operand order repaired and checked by reading, not compiled, not run on hardware
//
#include <hip/hip_runtime.h>
#include <hip/hip_bf16.h>
#include <stdint.h>
#include <math.h>

typedef _Float16 f16;
typedef __attribute__((ext_vector_type(16))) _Float16 v16h;
typedef __attribute__((ext_vector_type(8)))  _Float16 v8h;
typedef __attribute__((ext_vector_type(8)))  float    v8f;
typedef __attribute__((ext_vector_type(4)))  int      v4i_;

#define B_   64
#define T_   512
#define D_   300
#define H_   400
#define NL_  48
#define BT_  (B_*T_)        // 32768 rows
#define G4H  (4*H_)         // 1600
#define KP0  320            // D padded to 32-multiple
#define KPH  416            // H padded to 32-multiple
#define LD1  (2*H_)         // 800
#define NTH  25             // H/16 n-tiles per gate
#define KTH  13             // KPH/32 k-steps for recurrent GEMM

// ---- CDNA5 async LDS->global path (guarded; falls back to vector stores) ----
#if defined(__has_builtin)
#  if __has_builtin(__builtin_amdgcn_global_store_async_from_lds_b128)
#    define HAVE_ASYNC 1
#  endif
#endif
#ifndef HAVE_ASYNC
#  define HAVE_ASYNC 0
#endif

__device__ __forceinline__ void wait_async0(){
#if HAVE_ASYNC
#  if __has_builtin(__builtin_amdgcn_s_wait_asynccnt)
  __builtin_amdgcn_s_wait_asynccnt(0);
#  else
  asm volatile("s_wait_asynccnt 0x0" ::: "memory");
#  endif
#endif
}

__device__ __forceinline__ float sigm(float x){ return 1.f/(1.f+__expf(-x)); }

__device__ __forceinline__ v16h cat16(v8h lo, v8h hi){
  return __builtin_shufflevector(lo, hi, 0,1,2,3,4,5,6,7,8,9,10,11,12,13,14,15);
}
__device__ __forceinline__ v16h ldfragA(const f16* row, int k0){
  v8h lo = *(const v8h*)(row + k0);
  v8h hi = *(const v8h*)(row + k0 + 16);
  return cat16(lo, hi);
}
__device__ __forceinline__ v16h ldfragB(const f16* p){
  v8h lo = *(const v8h*)p;
  v8h hi = *(const v8h*)(p + 8);
  return cat16(lo, hi);
}
#define WMMA(Aop, Bop, Cop) \
  __builtin_amdgcn_wmma_f32_16x16x32_f16(false, (Aop), false, (Bop), (short)0, (Cop), false, false)

// ---------------- utility kernels ----------------
__global__ void k_zero(uint32_t* p, size_t n){
  size_t i = (size_t)blockIdx.x*blockDim.x + threadIdx.x;
  if (i < n) p[i] = 0u;
}

__global__ void k_pack_x(const float* __restrict__ x, f16* __restrict__ dst){
  size_t i = (size_t)blockIdx.x*blockDim.x + threadIdx.x;
  if (i >= (size_t)BT_*KP0) return;
  int row = (int)(i / KP0), k = (int)(i % KP0);
  float v = (k < D_) ? x[(size_t)row*D_ + k] : 0.f;
  dst[i] = (f16)v;
}

// pack W[row0+K][N] (fp32, ld=ldw) into WMMA-B fragment layout:
// linear = ((kt*nt + ntile)*32 + lane)*16 + e ; K = kt*32 + (lane/16)*8 + (e&7) + (e>>3)*16 ; N = ntile*16 + lane%16
__global__ void k_pack_w(const float* __restrict__ W, int ldw, int row0, int ksrc,
                         int ksteps, int nt, f16* __restrict__ dst){
  size_t i = (size_t)blockIdx.x*blockDim.x + threadIdx.x;
  size_t total = (size_t)ksteps*nt*512;
  if (i >= total) return;
  int e = (int)(i & 15);
  int l = (int)((i >> 4) & 31);
  int f = (int)(i >> 9);
  int ntile = f % nt, kt = f / nt;
  int K = kt*32 + (l>>4)*8 + (e&7) + (e>>3)*16;
  int N = ntile*16 + (l&15);
  float v = (K < ksrc) ? W[(size_t)(row0+K)*ldw + N] : 0.f;
  dst[i] = (f16)v;
}

// ---------------- big parallel GEMM: pregates = X @ Wx + b ----------------
// Register-blocked: each wave owns 1 m-tile x 4 n-tiles; one A fragment feeds
// 4 WMMAs per k-step. All 5 fragment loads issue before the WMMA cluster so
// the loads pipeline (distinct dest VGPRs -> staggered s_wait_loadcnt).
__global__ __launch_bounds__(256)
void k_xgemm(const f16* __restrict__ A, int lda, int ksteps,
             const f16* __restrict__ Bp, int nt,
             const float* __restrict__ bias, f16* __restrict__ out, int ldo){
  int wave = threadIdx.x >> 5, lane = threadIdx.x & 31;
  int li = lane & 15, hs = lane >> 4;
  int ng = nt >> 2;                        // groups of 4 n-tiles
  long wtile  = (long)blockIdx.x*8 + wave;
  long mt     = wtile / ng;
  int  ngi    = (int)(wtile % ng);
  long mbase  = mt*16;
  int  ntile0 = ngi*4;

  v8f acc[4];
  #pragma unroll
  for (int j = 0; j < 4; ++j){
    float bv = bias[(ntile0 + j)*16 + li];
    #pragma unroll
    for (int r = 0; r < 8; ++r) acc[j][r] = bv;
  }

  const f16* arow = A + (mbase + li)*(long)lda;
  const f16* bcol = Bp + ((long)ntile0*32 + lane)*16;
  long bstride = (long)nt*512;
  for (int kt = 0; kt < ksteps; ++kt){
    __builtin_prefetch(bcol + bstride, 0, 1);       // next k-step's B fragments
    v16h Af = ldfragA(arow, kt*32 + hs*8);
    v16h B0 = ldfragB(bcol);
    v16h B1 = ldfragB(bcol + 512);
    v16h B2 = ldfragB(bcol + 1024);
    v16h B3 = ldfragB(bcol + 1536);
    acc[0] = WMMA(Af, B0, acc[0]);
    acc[1] = WMMA(Af, B1, acc[1]);
    acc[2] = WMMA(Af, B2, acc[2]);
    acc[3] = WMMA(Af, B3, acc[3]);
    bcol += bstride;
  }
  #pragma unroll
  for (int j = 0; j < 4; ++j)
    #pragma unroll
    for (int r = 0; r < 8; ++r){
      long row = mbase + r + 8*hs;
      out[row*(long)ldo + (ntile0 + j)*16 + li] = (f16)acc[j][r];
    }
}

// ---------------- sequential bidirectional LSTM layer ----------------
// grid = 2 blocks (dir 0 = fw, 1 = bw), 1024 threads (32 waves) per block.
// h lives in LDS (WMMA A operand), c lives in registers, Wh streams from L2,
// per-step h output streams to global via async LDS->memory DMA (ASYNCcnt).
__global__ __launch_bounds__(1024)
void k_lstm_seq(const f16* __restrict__ pg_fw, const f16* __restrict__ pg_bw,
                const f16* __restrict__ Wh_fw, const f16* __restrict__ Wh_bw,
                const int* __restrict__ lens, f16* __restrict__ xnext, int ldn){
  __shared__ __align__(16) f16 h_sh[B_][KPH];
  __shared__ int len_sh[B_];

  int dir = blockIdx.x;
  const f16* pg = dir ? pg_bw : pg_fw;
  const f16* Wh = dir ? Wh_bw : Wh_fw;

  int tid = threadIdx.x, wave = tid >> 5, lane = tid & 31;
  int li = lane & 15, hs = lane >> 4;

  for (int i = tid; i < B_*KPH; i += 1024) ((f16*)h_sh)[i] = (f16)0.f;
  if (tid < B_) len_sh[tid] = lens[tid];
  __syncthreads();

  float c_reg[4][8];
  #pragma unroll
  for (int i = 0; i < 4; ++i)
    #pragma unroll
    for (int r = 0; r < 8; ++r) c_reg[i][r] = 0.f;

  for (int s = 0; s < T_; ++s){
    f16 hstash[4][8];
    #pragma unroll
    for (int i = 0; i < 4; ++i){
      int g = wave + 32*i;
      if (g < 100){
        int mt = g / NTH, nt = g % NTH;
        v8f acc[4];
        // init accumulators from precomputed x-pregates (+bias, already folded)
        #pragma unroll
        for (int q = 0; q < 4; ++q)
          #pragma unroll
          for (int r = 0; r < 8; ++r){
            int b   = mt*16 + r + 8*hs;
            int len = len_sh[b];
            int tv  = dir ? max(len - 1 - s, 0) : s;
            acc[q][r] = (float)pg[((long)b*T_ + tv)*G4H + q*H_ + nt*16 + li];
          }
        // recurrent GEMM: z += h @ Wh  (four gate column blocks per group)
        const f16* wb = Wh + ((long)nt*32 + lane)*16;
        for (int kt = 0; kt < KTH; ++kt){
          v16h Af = ldfragA(&h_sh[mt*16 + li][0], kt*32 + hs*8);
          v16h B0 = ldfragB(wb + (long)(0*NTH)*512);
          v16h B1 = ldfragB(wb + (long)(1*NTH)*512);
          v16h B2 = ldfragB(wb + (long)(2*NTH)*512);
          v16h B3 = ldfragB(wb + (long)(3*NTH)*512);
          acc[0] = WMMA(Af, B0, acc[0]);
          acc[1] = WMMA(Af, B1, acc[1]);
          acc[2] = WMMA(Af, B2, acc[2]);
          acc[3] = WMMA(Af, B3, acc[3]);
          wb += (long)100*512;
        }
        // elementwise LSTM cell (register-local: i/j/f/o share (row, unit) per lane)
        #pragma unroll
        for (int r = 0; r < 8; ++r){
          int b = mt*16 + r + 8*hs;
          float cn = c_reg[i][r]*sigm(acc[2][r] + 1.0f)   // forget_bias = 1.0
                   + sigm(acc[0][r])*tanhf(acc[1][r]);
          float hn = tanhf(cn)*sigm(acc[3][r]);
          if (s < len_sh[b]) c_reg[i][r] = cn;
          hstash[i][r] = (f16)hn;
        }
      }
    }
    // previous step's async stores must finish reading h_sh before we overwrite
    wait_async0();
    __syncthreads();
    #pragma unroll
    for (int i = 0; i < 4; ++i){
      int g = wave + 32*i;
      if (g < 100){
        int mt = g / NTH, nt = g % NTH;
        #pragma unroll
        for (int r = 0; r < 8; ++r){
          int b = mt*16 + r + 8*hs;
          if (s < len_sh[b]) h_sh[b][nt*16 + li] = hstash[i][r];
        }
      }
    }
    __syncthreads();   // new h visible to all waves
    // stream h(t) rows out of LDS: 64 rows x 50 x 16B chunks
    for (int c = tid; c < B_*(H_/8); c += 1024){
      int b = c / (H_/8), j = c % (H_/8);
      int len = len_sh[b];
      if (s < len){
        int tv = dir ? (len - 1 - s) : s;
        f16* gp = xnext + ((long)b*T_ + tv)*ldn + dir*H_ + j*8;
#if HAVE_ASYNC
        __builtin_amdgcn_global_store_async_from_lds_b128(
            (v4i_*)gp, (v4i_*)&h_sh[b][j*8], 0, 0);
#else
        *(v8h*)gp = *(const v8h*)&h_sh[b][j*8];
#endif
      }
    }
  }
}

// ---------------- batch-norm statistics, folded into scale/shift ----------------
__global__ void k_bnstats(const f16* __restrict__ x, const float* __restrict__ gamma,
                          const float* __restrict__ beta,
                          float* __restrict__ bn_s, float* __restrict__ bn_t){
  __shared__ float s1[256], s2[256];
  int c = blockIdx.x;
  float a = 0.f, b = 0.f;
  for (int r = threadIdx.x; r < BT_; r += 256){
    float v = (float)x[(long)r*LD1 + c];
    a += v; b += v*v;
  }
  s1[threadIdx.x] = a; s2[threadIdx.x] = b;
  __syncthreads();
  for (int off = 128; off; off >>= 1){
    if (threadIdx.x < off){
      s1[threadIdx.x] += s1[threadIdx.x + off];
      s2[threadIdx.x] += s2[threadIdx.x + off];
    }
    __syncthreads();
  }
  if (threadIdx.x == 0){
    float mean = s1[0] / (float)BT_;
    float var  = s2[0] / (float)BT_ - mean*mean;
    float sc   = gamma[c] * rsqrtf(var + 1e-3f);
    bn_s[c] = sc;
    bn_t[c] = beta[c] - mean*sc;
  }
}

__global__ void k_pack_wout(const float* __restrict__ W, const float* __restrict__ bn_s,
                            f16* __restrict__ dst){
  int i = blockIdx.x*blockDim.x + threadIdx.x;
  if (i >= 25*3*512) return;
  int e = i & 15, l = (i >> 4) & 31, f = i >> 9;
  int ntile = f % 3, kt = f / 3;
  int K = kt*32 + (l>>4)*8 + (e&7) + (e>>3)*16;
  int N = ntile*16 + (l&15);
  dst[i] = (f16)(W[(long)K*NL_ + N] * bn_s[K]);
}

__global__ void k_beff(const float* __restrict__ W, const float* __restrict__ bn_t,
                       const float* __restrict__ b_out, float* __restrict__ beff){
  int n = threadIdx.x;
  if (n >= NL_) return;
  float a = b_out[n];
  for (int k = 0; k < LD1; ++k) a += bn_t[k]*W[(long)k*NL_ + n];
  beff[n] = a;
}

// ---------------- projection + log-softmax + masked NLL ----------------
__global__ __launch_bounds__(32)
void k_loss(const f16* __restrict__ x, const f16* __restrict__ Wp,
            const float* __restrict__ beff, const int* __restrict__ tags,
            const float* __restrict__ mask, float* __restrict__ accum){
  int lane = threadIdx.x, li = lane & 15, hs = lane >> 4;
  long mbase = (long)blockIdx.x*16;

  v8f acc[3];
  #pragma unroll
  for (int q = 0; q < 3; ++q){
    float bv = beff[q*16 + li];
    #pragma unroll
    for (int r = 0; r < 8; ++r) acc[q][r] = bv;
  }
  const f16* arow = x + (mbase + li)*(long)LD1;
  const f16* wb = Wp + ((long)lane)*16;
  for (int kt = 0; kt < 25; ++kt){
    v16h Af = ldfragA(arow, kt*32 + hs*8);
    v16h B0 = ldfragB(wb);
    v16h B1 = ldfragB(wb + 512);
    v16h B2 = ldfragB(wb + 1024);
    acc[0] = WMMA(Af, B0, acc[0]);
    acc[1] = WMMA(Af, B1, acc[1]);
    acc[2] = WMMA(Af, B2, acc[2]);
    wb += 3*512;
  }
  #pragma unroll
  for (int r = 0; r < 8; ++r){
    long m = mbase + r + 8*hs;
    float l0 = acc[0][r], l1 = acc[1][r], l2 = acc[2][r];
    float lm = fmaxf(l0, fmaxf(l1, l2));
    for (int off = 8; off; off >>= 1) lm = fmaxf(lm, __shfl_xor(lm, off, 16));
    float z = __expf(l0-lm) + __expf(l1-lm) + __expf(l2-lm);
    for (int off = 8; off; off >>= 1) z += __shfl_xor(z, off, 16);
    int tg = tags[m];
    float lt = (li == tg ? l0 : 0.f) + (16+li == tg ? l1 : 0.f) + (32+li == tg ? l2 : 0.f);
    for (int off = 8; off; off >>= 1) lt += __shfl_xor(lt, off, 16);
    float nll = -(lt - lm - __logf(z));
    if (li == 0) atomicAdd(accum, nll * mask[m]);
  }
}

__global__ void k_finalize(const float* __restrict__ accum, const int* __restrict__ lens,
                           float* __restrict__ out){
  if (threadIdx.x == 0 && blockIdx.x == 0){
    int s = 0;
    for (int i = 0; i < B_; ++i) s += lens[i];
    out[0] = accum[0] / (float)s;
  }
}

// ---------------- host orchestration ----------------
extern "C" void kernel_launch(void* const* d_in, const int* in_sizes, int n_in,
                              void* d_out, int out_size, void* d_ws, size_t ws_size,
                              hipStream_t stream){
  (void)in_sizes; (void)n_in; (void)out_size;
  const float* inputs = (const float*)d_in[0];
  const int*   tags   = (const int*)  d_in[1];
  const float* mask   = (const float*)d_in[2];
  const int*   lens   = (const int*)  d_in[3];
  const float* W_fw0  = (const float*)d_in[4];
  const float* b_fw0  = (const float*)d_in[5];
  const float* W_bw0  = (const float*)d_in[6];
  const float* b_bw0  = (const float*)d_in[7];
  const float* W_fw1  = (const float*)d_in[8];
  const float* b_fw1  = (const float*)d_in[9];
  const float* W_bw1  = (const float*)d_in[10];
  const float* b_bw1  = (const float*)d_in[11];
  const float* gamma  = (const float*)d_in[12];
  const float* beta   = (const float*)d_in[13];
  const float* W_out  = (const float*)d_in[14];
  const float* b_out  = (const float*)d_in[15];

  char* ws = (char*)d_ws;
  size_t off = 0;
  auto alloc = [&](size_t bytes)->size_t{
    size_t o = off; off = (off + bytes + 255) & ~(size_t)255; return o;
  };
  size_t o_x0   = alloc((size_t)BT_*KP0*2);
  size_t o_x1   = alloc((size_t)BT_*LD1*2);
  size_t o_x2   = alloc((size_t)BT_*LD1*2);
  size_t o_pgf  = alloc((size_t)BT_*G4H*2);
  size_t o_pgb  = alloc((size_t)BT_*G4H*2);
  size_t o_wx0f = alloc((size_t)10*100*512*2);
  size_t o_wx0b = alloc((size_t)10*100*512*2);
  size_t o_wh0f = alloc((size_t)KTH*100*512*2);
  size_t o_wh0b = alloc((size_t)KTH*100*512*2);
  size_t o_wx1f = alloc((size_t)25*100*512*2);
  size_t o_wx1b = alloc((size_t)25*100*512*2);
  size_t o_wh1f = alloc((size_t)KTH*100*512*2);
  size_t o_wh1b = alloc((size_t)KTH*100*512*2);
  size_t o_wop  = alloc((size_t)25*3*512*2);
  size_t o_bns  = alloc((size_t)LD1*4);
  size_t o_bnt  = alloc((size_t)LD1*4);
  size_t o_beff = alloc((size_t)NL_*4);
  size_t o_acc  = alloc(4);
  if (ws_size < off) return;

  f16* x0  = (f16*)(ws + o_x0);
  f16* x1  = (f16*)(ws + o_x1);
  f16* x2  = (f16*)(ws + o_x2);
  f16* pgf = (f16*)(ws + o_pgf);
  f16* pgb = (f16*)(ws + o_pgb);
  f16* wx0f = (f16*)(ws + o_wx0f); f16* wx0b = (f16*)(ws + o_wx0b);
  f16* wh0f = (f16*)(ws + o_wh0f); f16* wh0b = (f16*)(ws + o_wh0b);
  f16* wx1f = (f16*)(ws + o_wx1f); f16* wx1b = (f16*)(ws + o_wx1b);
  f16* wh1f = (f16*)(ws + o_wh1f); f16* wh1b = (f16*)(ws + o_wh1b);
  f16* wop  = (f16*)(ws + o_wop);
  float* bns  = (float*)(ws + o_bns);
  float* bnt  = (float*)(ws + o_bnt);
  float* beff = (float*)(ws + o_beff);
  float* accum = (float*)(ws + o_acc);

  auto gz = [](size_t n){ return (unsigned)((n + 255)/256); };

  // zero x1/x2 (invalid timesteps must stay 0) and the loss accumulator
  size_t n32 = (size_t)BT_*LD1*2/4;
  k_zero<<<gz(n32), 256, 0, stream>>>((uint32_t*)x1, n32);
  k_zero<<<gz(n32), 256, 0, stream>>>((uint32_t*)x2, n32);
  k_zero<<<1, 256, 0, stream>>>((uint32_t*)accum, 1);

  // pack inputs + weights to f16 WMMA fragment layouts
  k_pack_x<<<gz((size_t)BT_*KP0), 256, 0, stream>>>(inputs, x0);
  k_pack_w<<<gz((size_t)10*100*512), 256, 0, stream>>>(W_fw0, G4H, 0,   D_, 10, 100, wx0f);
  k_pack_w<<<gz((size_t)10*100*512), 256, 0, stream>>>(W_bw0, G4H, 0,   D_, 10, 100, wx0b);
  k_pack_w<<<gz((size_t)KTH*100*512), 256, 0, stream>>>(W_fw0, G4H, D_,  H_, KTH, 100, wh0f);
  k_pack_w<<<gz((size_t)KTH*100*512), 256, 0, stream>>>(W_bw0, G4H, D_,  H_, KTH, 100, wh0b);
  k_pack_w<<<gz((size_t)25*100*512), 256, 0, stream>>>(W_fw1, G4H, 0,   LD1, 25, 100, wx1f);
  k_pack_w<<<gz((size_t)25*100*512), 256, 0, stream>>>(W_bw1, G4H, 0,   LD1, 25, 100, wx1b);
  k_pack_w<<<gz((size_t)KTH*100*512), 256, 0, stream>>>(W_fw1, G4H, LD1, H_, KTH, 100, wh1f);
  k_pack_w<<<gz((size_t)KTH*100*512), 256, 0, stream>>>(W_bw1, G4H, LD1, H_, KTH, 100, wh1b);

  // 2048 m-tiles * 25 n-groups (4 tiles each) / 8 waves per block
  const unsigned gemm_blocks = (2048u*25u)/8u;
  // layer 0: pregates, then sequential recurrence (both directions concurrently)
  k_xgemm<<<gemm_blocks, 256, 0, stream>>>(x0, KP0, 10, wx0f, 100, b_fw0, pgf, G4H);
  k_xgemm<<<gemm_blocks, 256, 0, stream>>>(x0, KP0, 10, wx0b, 100, b_bw0, pgb, G4H);
  k_lstm_seq<<<2, 1024, 0, stream>>>(pgf, pgb, wh0f, wh0b, lens, x1, LD1);
  // layer 1
  k_xgemm<<<gemm_blocks, 256, 0, stream>>>(x1, LD1, 25, wx1f, 100, b_fw1, pgf, G4H);
  k_xgemm<<<gemm_blocks, 256, 0, stream>>>(x1, LD1, 25, wx1b, 100, b_bw1, pgb, G4H);
  k_lstm_seq<<<2, 1024, 0, stream>>>(pgf, pgb, wh1f, wh1b, lens, x2, LD1);

  // batch-norm folded into projection, then fused projection + log-softmax + NLL
  k_bnstats<<<LD1, 256, 0, stream>>>(x2, gamma, beta, bns, bnt);
  k_pack_wout<<<gz((size_t)25*3*512), 256, 0, stream>>>(W_out, bns, wop);
  k_beff<<<1, 64, 0, stream>>>(W_out, bnt, b_out, beff);
  k_loss<<<2048, 32, 0, stream>>>(x2, wop, beff, tags, mask, accum);
  k_finalize<<<1, 1, 0, stream>>>(accum, lens, (float*)d_out);
}
